// MultiGraphConv_13975823581509
// MI455X (gfx1250) — matrix operand order repaired and checked by reading, not compiled, not run on hardware
//
#include <hip/hip_runtime.h>
#include <hip/hip_bf16.h>
#include <hip/hip_fp16.h>

#define B_ 2
#define N_ 2048
#define F_ 128
#define M_ 8
#define U_ 64

typedef __attribute__((ext_vector_type(16))) __bf16 v16bf;
typedef __attribute__((ext_vector_type(8)))  __bf16 bf16x8;
typedef __attribute__((ext_vector_type(8)))  float  v8f;

// D = A*B + C, with optional reuse-A hint (legal: next instr is identical op
// sharing the same A operand).
#define WMMA_BF16(Av, Bv, Cv, RA) \
  __builtin_amdgcn_wmma_f32_16x16x32_bf16(false, (Av), false, (Bv), (short)0, (Cv), (RA), false)

static __device__ __forceinline__ v16bf cat8(bf16x8 lo, bf16x8 hi) {
  return __builtin_shufflevector(lo, hi, 0, 1, 2, 3, 4, 5, 6, 7,
                                         8, 9, 10, 11, 12, 13, 14, 15);
}

// --------------------------------------------------------------------------
// Kernel 1: alpha[b,n,m,u] = sum_f seq[b,n,f] * vw[f,m,u]  -> written to d_out
// (pre-sigmoid bias; the main kernel adds the graph GEMM and applies sigmoid)
// --------------------------------------------------------------------------
__global__ void __launch_bounds__(M_*U_)
alpha_kernel(const float* __restrict__ seq,
             const float* __restrict__ vw,
             float* __restrict__ out) {
  const int mu  = threadIdx.x;                 // 0..511  (m*64+u)
  const int blk = blockIdx.x;                  // B_*N_/8 blocks
  const int b   = blk / (N_ / 8);
  const int n0  = (blk % (N_ / 8)) * 8;
  const float* srow = seq + ((size_t)b * N_ + n0) * F_;
  float acc[8] = {0.f, 0.f, 0.f, 0.f, 0.f, 0.f, 0.f, 0.f};
  for (int f = 0; f < F_; ++f) {
    const float w = vw[(size_t)f * (M_ * U_) + mu];
#pragma unroll
    for (int c = 0; c < 8; ++c)
      acc[c] = __builtin_fmaf(srow[(size_t)c * F_ + f], w, acc[c]);
  }
#pragma unroll
  for (int c = 0; c < 8; ++c)
    out[((size_t)b * N_ + n0 + c) * (M_ * U_) + mu] = acc[c];
}

// --------------------------------------------------------------------------
// Kernel 2: betaT[b][u][n] = (bf16) sum_f seq[b,n,f] * ew[f,u]
// Transposed (u-major) bf16 layout: each main-kernel B fragment is one
// 32B-contiguous, 32B-aligned run per lane in the main kernel's K = khalf*16 + e order.
// --------------------------------------------------------------------------
__global__ void __launch_bounds__(256)
beta_kernel(const float* __restrict__ seq,
            const float* __restrict__ ew,
            __bf16* __restrict__ betaT) {
  const int tid = threadIdx.x;                 // 256
  const int u   = tid & (U_ - 1);
  const int c   = tid >> 6;                    // 0..3
  const int blk = blockIdx.x;                  // B_*N_/4 blocks
  const int b   = blk / (N_ / 4);
  const int n   = (blk % (N_ / 4)) * 4 + c;
  const float* srow = seq + ((size_t)b * N_ + n) * F_;
  float acc = 0.f;
  for (int f = 0; f < F_; ++f)
    acc = __builtin_fmaf(srow[f], ew[(size_t)f * U_ + u], acc);
  betaT[((size_t)b * U_ + u) * N_ + n] = (__bf16)acc;
}

// --------------------------------------------------------------------------
// Main kernel: one wave per (b, i-pair). Computes the 16x64 tile
//   C[(di*8+m), u] = sum_j graph[b, i0+di, j, m] * beta[b, j, u]
// via v_wmma_f32_16x16x32_bf16 with software-pipelined (double-buffered)
// A and B fragments, then out = sigmoid(C + alpha) (alpha staged in d_out).
//
// A-fragment (16x32 bf16) lane map: lane L: row r=L&15 (di=r>>3, m=r&7),
// khalf=L>>4; element e holds K = khalf*16 + e.
// B-fragment (32x16 bf16) lane map: lane L: col n = utile*16 + (L&15),
// element e holds K = khalf*16 + e  ->  16 consecutive bf16 from betaT.
// --------------------------------------------------------------------------
__global__ void __launch_bounds__(32)
mgc_kernel(const float* __restrict__ graph,
           const __bf16* __restrict__ betaT,
           float* __restrict__ out) {
  const int L   = threadIdx.x;                 // 0..31
  const int r   = L & 15;
  const int kh  = L >> 4;                      // 0 or 1
  const int di  = r >> 3;
  const int m   = r & 7;
  const int blk = blockIdx.x;                  // B_*N_/2 blocks
  const int b   = blk / (N_ / 2);
  const int i0  = (blk % (N_ / 2)) * 2;

  // per-lane A row base: graph[b, i0+di, j, m]; j-stride is M_ floats (32B)
  const float* ap = graph + ((size_t)b * N_ + (i0 + di)) * (size_t)N_ * M_ + m;

  // per-lane B bases for the 4 u-tiles (named scalars -> global addrspace)
  const bf16x8* bp0 = (const bf16x8*)(betaT + ((size_t)b * U_ +  0 + r) * N_ + kh * 16);
  const bf16x8* bp1 = (const bf16x8*)(betaT + ((size_t)b * U_ + 16 + r) * N_ + kh * 16);
  const bf16x8* bp2 = (const bf16x8*)(betaT + ((size_t)b * U_ + 32 + r) * N_ + kh * 16);
  const bf16x8* bp3 = (const bf16x8*)(betaT + ((size_t)b * U_ + 48 + r) * N_ + kh * 16);

  auto load_a = [&](const float* p) -> v16bf {
    v16bf a;
#pragma unroll
    for (int e = 0; e < 16; ++e) {
      const int K = e + (kh << 4);   // K in 0..31
      a[e] = (__bf16)__builtin_nontemporal_load(p + (size_t)K * M_);
    }
    return a;
  };

  v8f acc0 = {}, acc1 = {}, acc2 = {}, acc3 = {};

  // ---- prologue: fragments for k-step 0 ----
  v16bf a_cur = load_a(ap);
  bf16x8 lo0 = bp0[0], hi0 = bp0[1];
  bf16x8 lo1 = bp1[0], hi1 = bp1[1];
  bf16x8 lo2 = bp2[0], hi2 = bp2[1];
  bf16x8 lo3 = bp3[0], hi3 = bp3[1];

  // ---- steady state: issue next tile's loads, then this tile's WMMAs ----
  for (int it = 0; it < (N_ / 32) - 1; ++it) {
    ap  += 32 * M_;                 // next 32 j's
    bp0 += 4; bp1 += 4; bp2 += 4; bp3 += 4;   // +32 bf16 elements

    const v16bf a_nxt = load_a(ap);
    const bf16x8 nlo0 = bp0[0], nhi0 = bp0[1];
    const bf16x8 nlo1 = bp1[0], nhi1 = bp1[1];
    const bf16x8 nlo2 = bp2[0], nhi2 = bp2[1];
    const bf16x8 nlo3 = bp3[0], nhi3 = bp3[1];
    __builtin_prefetch(ap + 32 * M_, 0, 1);   // graph stream, one tile ahead

    acc0 = WMMA_BF16(a_cur, cat8(lo0, hi0), acc0, true);
    acc1 = WMMA_BF16(a_cur, cat8(lo1, hi1), acc1, true);
    acc2 = WMMA_BF16(a_cur, cat8(lo2, hi2), acc2, true);
    acc3 = WMMA_BF16(a_cur, cat8(lo3, hi3), acc3, false);

    a_cur = a_nxt;
    lo0 = nlo0; hi0 = nhi0;  lo1 = nlo1; hi1 = nhi1;
    lo2 = nlo2; hi2 = nhi2;  lo3 = nlo3; hi3 = nhi3;
  }

  // ---- tail: last k-step ----
  acc0 = WMMA_BF16(a_cur, cat8(lo0, hi0), acc0, true);
  acc1 = WMMA_BF16(a_cur, cat8(lo1, hi1), acc1, true);
  acc2 = WMMA_BF16(a_cur, cat8(lo2, hi2), acc2, true);
  acc3 = WMMA_BF16(a_cur, cat8(lo3, hi3), acc3, false);

  // Epilogue: C/D f32 layout: VGPR q, lanes 0-15 -> row q, lanes 16-31 -> row q+8
  v8f accs[4] = {acc0, acc1, acc2, acc3};
#pragma unroll
  for (int q = 0; q < 8; ++q) {
    const int rr  = q + (kh << 3);
    const int dio = rr >> 3;
    const int mo  = rr & 7;
    const size_t base = (((size_t)b * N_ + i0 + dio) * M_ + mo) * U_;
#pragma unroll
    for (int t = 0; t < 4; ++t) {
      const size_t oidx = base + t * 16 + r;
      const float v = accs[t][q] + out[oidx];  // alpha staged in d_out
      out[oidx] = 1.0f / (1.0f + __expf(-v));
    }
  }
}

// --------------------------------------------------------------------------
extern "C" void kernel_launch(void* const* d_in, const int* in_sizes, int n_in,
                              void* d_out, int out_size, void* d_ws, size_t ws_size,
                              hipStream_t stream) {
  const float* seq   = (const float*)d_in[0];   // (B,N,F)
  const float* graph = (const float*)d_in[1];   // (B,N,N,M)
  const float* ew    = (const float*)d_in[2];   // (F,U)
  const float* vw    = (const float*)d_in[3];   // (F,M,U)
  float* out = (float*)d_out;                   // (B,N,M,U)

  __bf16* betaT = (__bf16*)d_ws;                // B_*U_*N_ bf16 = 512 KB

  alpha_kernel<<<B_ * N_ / 8, M_ * U_, 0, stream>>>(seq, vw, out);
  beta_kernel <<<B_ * N_ / 4, 256,     0, stream>>>(seq, ew, betaT);
  mgc_kernel  <<<B_ * N_ / 2, 32,      0, stream>>>(graph, betaT, out);
}
